// TransformerBlock_81329500717162
// MI455X (gfx1250) — compile-verified, matrix operand order checked by reading, not run on hardware
//
#include <hip/hip_runtime.h>
#include <hip/hip_bf16.h>
#include <math.h>

typedef _Float16 f16;
typedef __attribute__((ext_vector_type(16))) _Float16 v16h;
typedef __attribute__((ext_vector_type(8)))  _Float16 v8h;
typedef __attribute__((ext_vector_type(8)))  float    v8f;

#define B_   8
#define T_   1024
#define D_   1024
#define H_   16
#define HS_  64
#define FF_  4096
#define M_   (B_ * T_)   // 8192

// ---------------------------------------------------------------------------
// WMMA helpers. CDNA5 16-bit operand layout (ISA 7.12.2):
//  A (16x32, MxK): lane lo (=lane&15) holds row M=lo; element e in [0,8):
//    K = hi*8 + e ; e in [8,16): K = 16 + hi*8 + (e-8)   (hi = lane>>4)
//  -> per lane two contiguous 8-half (16B) runs: perfect for b128 loads.
//  B (32x16, KxN) mirrors A with N=lo when read from B^T ([N][K] row-major).
//  C/D (16x16 f32): lane lo = col N; VGPR i = row M = i + hi*8.
// ---------------------------------------------------------------------------
__device__ __forceinline__ v16h load_tile16x32(const f16* __restrict__ p, int ld, int lane) {
  const int lo = lane & 15;
  const int hi = (lane >> 4) & 1;
  const f16* r = p + (size_t)lo * ld + hi * 8;
  v8h a0 = *(const v8h*)(r);
  v8h a1 = *(const v8h*)(r + 16);
  v16h out;
#pragma unroll
  for (int e = 0; e < 8; ++e) { out[e] = a0[e]; out[e + 8] = a1[e]; }
  return out;
}

__device__ __forceinline__ v8f wmma_f16(v16h a, v16h b, v8f c) {
  // (neg_a, A, neg_b, B, c_mod, C, reuse_a, reuse_b)
  return __builtin_amdgcn_wmma_f32_16x16x32_f16(false, a, false, b, (short)0, c,
                                                false, false);
}

__device__ __forceinline__ v8f v8f_zero() {
  v8f z = {0.f, 0.f, 0.f, 0.f, 0.f, 0.f, 0.f, 0.f};
  return z;
}

// ---------------------------------------------------------------------------
// Double-buffered 4Mx2N GEMM core: acc[j*4+i] = (Mtile i, Ntile j).
// K must be a multiple of 64. Loads for the next 32-deep slice are issued
// before the current slice's WMMAs so the backend can emit partial
// s_wait_loadcnt and overlap VMEM with matrix math.
// ---------------------------------------------------------------------------
__device__ __forceinline__ void gemm_core_4x2(const f16* __restrict__ A,
                                              const f16* __restrict__ BT,
                                              int K, int m0, int n0, int lane,
                                              v8f acc[8]) {
#pragma unroll
  for (int i = 0; i < 8; ++i) acc[i] = v8f_zero();

  const f16* Arow = A + (size_t)m0 * K;
  const f16* Brow = BT + (size_t)n0 * K;
  const int lo = lane & 15;

  v16h aP[4], bP[2], aQ[4], bQ[2];
  bP[0] = load_tile16x32(Brow, K, lane);
  bP[1] = load_tile16x32(Brow + (size_t)16 * K, K, lane);
#pragma unroll
  for (int i = 0; i < 4; ++i)
    aP[i] = load_tile16x32(Arow + (size_t)(16 * i) * K, K, lane);

  for (int k0 = 0; k0 < K; k0 += 64) {
    const int kq = k0 + 32;
    bQ[0] = load_tile16x32(Brow + kq, K, lane);
    bQ[1] = load_tile16x32(Brow + (size_t)16 * K + kq, K, lane);
#pragma unroll
    for (int i = 0; i < 4; ++i)
      aQ[i] = load_tile16x32(Arow + (size_t)(16 * i) * K + kq, K, lane);

    if (k0 + 256 < K) {  // streaming prefetch (global_prefetch_b8)
      __builtin_prefetch(Brow + (size_t)lo * K + k0 + 256, 0, 1);
      __builtin_prefetch(Arow + (size_t)lo * K + k0 + 256, 0, 1);
    }

#pragma unroll
    for (int j = 0; j < 2; ++j)
#pragma unroll
      for (int i = 0; i < 4; ++i)
        acc[j * 4 + i] = wmma_f16(aP[i], bP[j], acc[j * 4 + i]);

    if (k0 + 64 < K) {
      const int kp = k0 + 64;
      bP[0] = load_tile16x32(Brow + kp, K, lane);
      bP[1] = load_tile16x32(Brow + (size_t)16 * K + kp, K, lane);
#pragma unroll
      for (int i = 0; i < 4; ++i)
        aP[i] = load_tile16x32(Arow + (size_t)(16 * i) * K + kp, K, lane);
    }

#pragma unroll
    for (int j = 0; j < 2; ++j)
#pragma unroll
      for (int i = 0; i < 4; ++i)
        acc[j * 4 + i] = wmma_f16(aQ[i], bQ[j], acc[j * 4 + i]);
  }
}

// ---------------------------------------------------------------------------
// Weight convert + transpose: src f32 [batch][R][C] -> dst f16 [batch][C][R]
// ---------------------------------------------------------------------------
__global__ __launch_bounds__(256) void convt_kernel(const float* __restrict__ src,
                                                    f16* __restrict__ dst,
                                                    int R, int C) {
  const size_t base = (size_t)blockIdx.z * R * C;
  const int idx = blockIdx.x * 256 + threadIdx.x;
  if (idx < R * C) {
    const int r = idx / C, c = idx % C;
    dst[base + (size_t)c * R + r] = (f16)src[base + idx];
  }
}

__global__ __launch_bounds__(256) void concat3_kernel(const float* __restrict__ a,
                                                      const float* __restrict__ b,
                                                      const float* __restrict__ c,
                                                      float* __restrict__ dst, int n) {
  const int i = blockIdx.x * 256 + threadIdx.x;
  if (i < 3 * n) dst[i] = (i < n) ? a[i] : ((i < 2 * n) ? b[i - n] : c[i - 2 * n]);
}

// ---------------------------------------------------------------------------
// LayerNorm (unbiased variance /(d-1), matching torch.var default)
// ---------------------------------------------------------------------------
__global__ __launch_bounds__(256) void ln_kernel(const float* __restrict__ x,
                                                 const float* __restrict__ gamma,
                                                 const float* __restrict__ beta,
                                                 float* __restrict__ y32,
                                                 f16* __restrict__ y16, int d) {
  __shared__ float red[256];
  const int row = blockIdx.x;
  const int tid = threadIdx.x;
  const float* xr = x + (size_t)row * d;

  float s = 0.f;
  for (int i = tid; i < d; i += 256) s += xr[i];
  red[tid] = s; __syncthreads();
  for (int off = 128; off > 0; off >>= 1) {
    if (tid < off) red[tid] += red[tid + off];
    __syncthreads();
  }
  const float mean = red[0] / (float)d;
  __syncthreads();

  float v = 0.f;
  for (int i = tid; i < d; i += 256) { float t = xr[i] - mean; v += t * t; }
  red[tid] = v; __syncthreads();
  for (int off = 128; off > 0; off >>= 1) {
    if (tid < off) red[tid] += red[tid + off];
    __syncthreads();
  }
  const float var  = red[0] / (float)(d - 1);
  const float rstd = rsqrtf(var + 1e-5f);

  for (int i = tid; i < d; i += 256) {
    float yv = (xr[i] - mean) * rstd * gamma[i] + beta[i];
    y32[(size_t)row * d + i] = yv;
    y16[(size_t)row * d + i] = (f16)yv;
  }
}

// ---------------------------------------------------------------------------
// Fused QKV GEMM: C[M=8192, N=3072] = xn16 * W^T + bias, routed to q/k/vT f16.
// Wave computes a 64x32 output tile (4Mx2N WMMA subtiles).
// ---------------------------------------------------------------------------
__global__ __launch_bounds__(128) void qkv_kernel(const f16* __restrict__ A,
                                                  const f16* __restrict__ BT,
                                                  const float* __restrict__ biascat,
                                                  f16* __restrict__ q,
                                                  f16* __restrict__ k,
                                                  f16* __restrict__ vT) {
  const int lane = threadIdx.x & 31;
  const int w    = threadIdx.x >> 5;
  const int lo   = lane & 15, hi = (lane >> 4) & 1;
  const int n0   = (blockIdx.x * 4 + w) * 32;
  const int m0   = blockIdx.y * 64;

  v8f acc[8];
  gemm_core_4x2(A, BT, D_, m0, n0, lane, acc);

#pragma unroll
  for (int j = 0; j < 2; ++j) {
    const int n  = n0 + j * 16 + lo;   // 0..3071
    const float bv = biascat[n];
    const int which = n >> 10;         // 0=q 1=k 2=v
    const int nn = n & 1023;
    const int hh = nn >> 6, e = nn & 63;
#pragma unroll
    for (int i = 0; i < 4; ++i) {
#pragma unroll
      for (int r = 0; r < 8; ++r) {
        const int row = m0 + 16 * i + r + hi * 8;   // global token index
        const int b = row >> 10, t = row & 1023;
        const float val = acc[j * 4 + i][r] + bv;
        const size_t bh = (size_t)(b * H_ + hh);
        if (which == 0)      q[(bh * T_ + t) * HS_ + e]  = (f16)val;
        else if (which == 1) k[(bh * T_ + t) * HS_ + e]  = (f16)val;
        else                 vT[(bh * HS_ + e) * T_ + t] = (f16)val;
      }
    }
  }
}

// ---------------------------------------------------------------------------
// Flash-style causal attention. One wave per 16-row q tile; j in 32-col tiles.
// ---------------------------------------------------------------------------
__global__ __launch_bounds__(128) void attn_kernel(const f16* __restrict__ q,
                                                   const f16* __restrict__ k,
                                                   const f16* __restrict__ vT,
                                                   f16* __restrict__ o) {
  __shared__ _Float16 plds[4][16][32];   // per-wave P staging (1KB each)
  const int lane = threadIdx.x & 31;
  const int w    = threadIdx.x >> 5;
  const int lo   = lane & 15, hi = (lane >> 4) & 1;
  const int bh   = blockIdx.y;
  const int i0   = blockIdx.x * 4 + w;   // 16-row q tile index (0..63)
  const int b    = bh >> 4, h = bh & 15;

  const size_t qbase = ((size_t)bh * T_ + i0 * 16) * HS_;
  const v16h qa0 = load_tile16x32(q + qbase, HS_, lane);
  const v16h qa1 = load_tile16x32(q + qbase + 32, HS_, lane);

  float m_run[8], l_run[8];
  v8f oacc[4];
#pragma unroll
  for (int i = 0; i < 8; ++i) { m_run[i] = -1e30f; l_run[i] = 0.f; }
#pragma unroll
  for (int t = 0; t < 4; ++t) oacc[t] = v8f_zero();

  const int rmax = i0 * 16 + 15;
  for (int j0 = 0; j0 <= rmax; j0 += 32) {
    v8f s0 = v8f_zero(), s1 = v8f_zero();
    {
      const size_t kb = ((size_t)bh * T_ + j0) * HS_;
      v16h t0 = load_tile16x32(k + kb, HS_, lane);
      s0 = wmma_f16(qa0, t0, s0);
      t0 = load_tile16x32(k + kb + 32, HS_, lane);
      s0 = wmma_f16(qa1, t0, s0);
      t0 = load_tile16x32(k + kb + 16 * HS_, HS_, lane);
      s1 = wmma_f16(qa0, t0, s1);
      t0 = load_tile16x32(k + kb + 16 * HS_ + 32, HS_, lane);
      s1 = wmma_f16(qa1, t0, s1);
    }
#pragma unroll
    for (int i = 0; i < 8; ++i) {
      const int row = i0 * 16 + i + hi * 8;
      float v0 = s0[i] * 0.125f;           // 1/sqrt(64)
      float v1 = s1[i] * 0.125f;
      if (j0 + lo > row)      v0 = -1e30f; // causal mask
      if (j0 + 16 + lo > row) v1 = -1e30f;
      float tm = fmaxf(v0, v1);
#pragma unroll
      for (int msk = 1; msk < 16; msk <<= 1) tm = fmaxf(tm, __shfl_xor(tm, msk, 32));
      const float mn    = fmaxf(m_run[i], tm);
      const float alpha = expf(m_run[i] - mn);
      const float p0 = expf(v0 - mn);
      const float p1 = expf(v1 - mn);
      float rs = p0 + p1;
#pragma unroll
      for (int msk = 1; msk < 16; msk <<= 1) rs += __shfl_xor(rs, msk, 32);
      l_run[i] = l_run[i] * alpha + rs;
      m_run[i] = mn;
#pragma unroll
      for (int t = 0; t < 4; ++t) oacc[t][i] *= alpha;
      plds[w][i + hi * 8][lo]      = (f16)p0;   // C-layout -> LDS
      plds[w][i + hi * 8][16 + lo] = (f16)p1;
    }
    asm volatile("s_wait_dscnt 0" ::: "memory");
    const v16h pa = load_tile16x32(&plds[w][0][0], 32, lane);  // A-layout read
#pragma unroll
    for (int t = 0; t < 4; ++t) {
      v16h vb = load_tile16x32(vT + ((size_t)bh * HS_ + t * 16) * T_ + j0, T_, lane);
      oacc[t] = wmma_f16(pa, vb, oacc[t]);
    }
    asm volatile("" ::: "memory");  // keep LDS reads before next iter's writes
  }

#pragma unroll
  for (int t = 0; t < 4; ++t) {
#pragma unroll
    for (int i = 0; i < 8; ++i) {
      const int row = i0 * 16 + i + hi * 8;
      const float val = oacc[t][i] / l_run[i];
      o[((size_t)b * T_ + row) * D_ + h * HS_ + t * 16 + lo] = (f16)val;
    }
  }
}

// ---------------------------------------------------------------------------
// GEMM + bias + residual (f32 out): out = A*B + bias + resid
// ---------------------------------------------------------------------------
__global__ __launch_bounds__(128) void gemm_bias_res_kernel(
    const f16* __restrict__ A, const f16* __restrict__ BT,
    const float* __restrict__ bias, const float* __restrict__ resid,
    float* __restrict__ out, int N, int K) {
  const int lane = threadIdx.x & 31;
  const int w    = threadIdx.x >> 5;
  const int lo   = lane & 15, hi = (lane >> 4) & 1;
  const int n0   = (blockIdx.x * 4 + w) * 32;
  const int m0   = blockIdx.y * 64;

  v8f acc[8];
  gemm_core_4x2(A, BT, K, m0, n0, lane, acc);

#pragma unroll
  for (int j = 0; j < 2; ++j) {
    const int n = n0 + j * 16 + lo;
    const float bv = bias[n];
#pragma unroll
    for (int i = 0; i < 4; ++i) {
#pragma unroll
      for (int r = 0; r < 8; ++r) {
        const int row = m0 + 16 * i + r + hi * 8;
        const size_t idx = (size_t)row * N + n;
        out[idx] = acc[j * 4 + i][r] + bv + resid[idx];
      }
    }
  }
}

// ---------------------------------------------------------------------------
// GEMM + bias + exact GELU (f16 out)
// ---------------------------------------------------------------------------
__global__ __launch_bounds__(128) void gemm_bias_gelu_kernel(
    const f16* __restrict__ A, const f16* __restrict__ BT,
    const float* __restrict__ bias, f16* __restrict__ out, int N, int K) {
  const int lane = threadIdx.x & 31;
  const int w    = threadIdx.x >> 5;
  const int lo   = lane & 15, hi = (lane >> 4) & 1;
  const int n0   = (blockIdx.x * 4 + w) * 32;
  const int m0   = blockIdx.y * 64;

  v8f acc[8];
  gemm_core_4x2(A, BT, K, m0, n0, lane, acc);

#pragma unroll
  for (int j = 0; j < 2; ++j) {
    const int n = n0 + j * 16 + lo;
    const float bv = bias[n];
#pragma unroll
    for (int i = 0; i < 4; ++i) {
#pragma unroll
      for (int r = 0; r < 8; ++r) {
        const int row = m0 + 16 * i + r + hi * 8;
        const float x = acc[j * 4 + i][r] + bv;
        const float g = 0.5f * x * (1.0f + erff(x * 0.70710678118654752f));
        out[(size_t)row * N + n] = (f16)g;
      }
    }
  }
}

// ---------------------------------------------------------------------------
// Host orchestration
// ---------------------------------------------------------------------------
extern "C" void kernel_launch(void* const* d_in, const int* in_sizes, int n_in,
                              void* d_out, int out_size, void* d_ws, size_t ws_size,
                              hipStream_t stream) {
  (void)in_sizes; (void)n_in; (void)out_size; (void)ws_size;

  const float* x      = (const float*)d_in[0];
  const float* wq     = (const float*)d_in[1];
  const float* wk     = (const float*)d_in[2];
  const float* wv     = (const float*)d_in[3];
  const float* bq     = (const float*)d_in[4];
  const float* bk     = (const float*)d_in[5];
  const float* bv     = (const float*)d_in[6];
  const float* w_proj = (const float*)d_in[7];
  const float* b_proj = (const float*)d_in[8];
  const float* w1     = (const float*)d_in[9];
  const float* b1     = (const float*)d_in[10];
  const float* w2     = (const float*)d_in[11];
  const float* b2     = (const float*)d_in[12];
  const float* gamma1 = (const float*)d_in[13];
  const float* beta1  = (const float*)d_in[14];
  const float* gamma2 = (const float*)d_in[15];
  const float* beta2  = (const float*)d_in[16];
  float* out = (float*)d_out;

  char* ws = (char*)d_ws;
  size_t off = 0;
  auto carve = [&](size_t bytes) -> char* {
    char* p = ws + off;
    off += (bytes + 255) & ~(size_t)255;
    return p;
  };
  float* xn32   = (float*)carve((size_t)M_ * D_ * 4);
  f16*   xn16   = (f16*)  carve((size_t)M_ * D_ * 2);
  float* x2_32  = (float*)carve((size_t)M_ * D_ * 4);
  float* x2n32  = (float*)carve((size_t)M_ * D_ * 4);
  f16*   x2n16  = (f16*)  carve((size_t)M_ * D_ * 2);
  f16*   q16    = (f16*)  carve((size_t)B_ * H_ * T_ * HS_ * 2);
  f16*   k16    = (f16*)  carve((size_t)B_ * H_ * T_ * HS_ * 2);
  f16*   vT16   = (f16*)  carve((size_t)B_ * H_ * T_ * HS_ * 2);
  f16*   o16    = (f16*)  carve((size_t)M_ * D_ * 2);
  f16*   h16    = (f16*)  carve((size_t)M_ * FF_ * 2);
  f16*   wqkvT  = (f16*)  carve((size_t)3 * D_ * D_ * 2);
  float* biasqkv= (float*)carve((size_t)3 * D_ * 4);
  f16*   wprojT = (f16*)  carve((size_t)D_ * D_ * 2);
  f16*   w1T    = (f16*)  carve((size_t)D_ * FF_ * 2);
  f16*   w2T    = (f16*)  carve((size_t)FF_ * D_ * 2);

  // --- weight conversion / transposition ---
  convt_kernel<<<dim3((D_ * HS_ + 255) / 256, 1, H_), 256, 0, stream>>>(
      wq, wqkvT + 0 * (size_t)D_ * D_, D_, HS_);
  convt_kernel<<<dim3((D_ * HS_ + 255) / 256, 1, H_), 256, 0, stream>>>(
      wk, wqkvT + 1 * (size_t)D_ * D_, D_, HS_);
  convt_kernel<<<dim3((D_ * HS_ + 255) / 256, 1, H_), 256, 0, stream>>>(
      wv, wqkvT + 2 * (size_t)D_ * D_, D_, HS_);
  convt_kernel<<<dim3((D_ * D_ + 255) / 256, 1, 1), 256, 0, stream>>>(
      w_proj, wprojT, D_, D_);
  convt_kernel<<<dim3((D_ * FF_ + 255) / 256, 1, 1), 256, 0, stream>>>(
      w1, w1T, D_, FF_);
  convt_kernel<<<dim3((FF_ * D_ + 255) / 256, 1, 1), 256, 0, stream>>>(
      w2, w2T, FF_, D_);
  concat3_kernel<<<(3 * D_ + 255) / 256, 256, 0, stream>>>(bq, bk, bv, biasqkv, D_);

  // --- LN1 ---
  ln_kernel<<<M_, 256, 0, stream>>>(x, gamma1, beta1, xn32, xn16, D_);

  // --- fused QKV GEMM (M=8192, N=3072, K=1024) ---
  qkv_kernel<<<dim3(3 * D_ / 128, M_ / 64), 128, 0, stream>>>(
      xn16, wqkvT, biasqkv, q16, k16, vT16);

  // --- attention ---
  attn_kernel<<<dim3(T_ / 64, B_ * H_), 128, 0, stream>>>(q16, k16, vT16, o16);

  // --- proj + bias + residual (M=8192, N=1024, K=1024) ---
  gemm_bias_res_kernel<<<dim3(D_ / 128, M_ / 64), 128, 0, stream>>>(
      o16, wprojT, b_proj, xn32, x2_32, D_, D_);

  // --- LN2 ---
  ln_kernel<<<M_, 256, 0, stream>>>(x2_32, gamma2, beta2, x2n32, x2n16, D_);

  // --- FFN1 + GELU (M=8192, N=4096, K=1024) ---
  gemm_bias_gelu_kernel<<<dim3(FF_ / 128, M_ / 64), 128, 0, stream>>>(
      x2n16, w1T, b1, h16, FF_, D_);

  // --- FFN2 + bias + residual -> out (M=8192, N=1024, K=4096) ---
  gemm_bias_res_kernel<<<dim3(D_ / 128, M_ / 64), 128, 0, stream>>>(
      h16, w2T, b2, x2n32, out, D_, FF_);
}
